// PN2DenseLatentEncoding_71880572666497
// MI455X (gfx1250) — matrix operand order repaired and checked by reading, not compiled, not run on hardware
//
#include <hip/hip_runtime.h>
#include <hip/hip_bf16.h>

// ---------------------------------------------------------------------------
// PN2DenseLatentEncoding forward for MI455X (gfx1250, wave32, WMMA).
// All MLP layers run through one f16-in/f32-acc WMMA GEMM kernel
// (v_wmma_f32_16x16x32_f16). Weights are pre-packed per launch into f16 in
// exact B-fragment order; activations are kept in f16 with padded row strides
// so the GEMM inner loop is branch-free: 6x global_load_b128 per 2 wmmas.
// ---------------------------------------------------------------------------

typedef _Float16 half_t;
typedef __attribute__((ext_vector_type(16))) _Float16 v16h;
typedef __attribute__((ext_vector_type(8)))  _Float16 v8h;
typedef __attribute__((ext_vector_type(8)))  float    v8f;

#define BDIM 256
#define FPS_MAXN 2048
#define KNBR 64

#define ACT_NONE  0
#define ACT_RELU  1
#define ACT_LEAKY 2

// ---------------------------------------------------------------------------
// WMMA GEMM:  Y[P,*] = act(X[P,Kpad] @ Wp + bias)
// One wave computes a 16x32 output tile (A fragment reused for two B tiles).
// Fragment layouts used here (f16 16x16x32; A/B in contiguous k, not the ISA table order):
//   A: lane l holds row M=l%16, half=l/16; halfs k= k0+16*half..+15
//      -> one contiguous 32B chunk of the f16 X row (matches B).
//   B: lane l holds col N=l%16; needs k'=16*half..16*half+15 of the k-tile
//      -> Wp packed as [Ktile][Npad][32] so that's one contiguous 32B chunk.
//   C/D: lane l col N=l%16; VGPR r row M = r + 8*half.
// Padded columns carry zero weights + zero bias -> compute exact zeros, so
// f16 stores are unguarded in N. Rows are clamped on load, guarded on store.
// ---------------------------------------------------------------------------
__global__ __launch_bounds__(32)
void wmma_gemm_kernel(const half_t* __restrict__ X, const half_t* __restrict__ Wp,
                      const float* __restrict__ bp, half_t* __restrict__ Yh,
                      float* __restrict__ Yf, int P, int Kpad, int Npad,
                      int Cout, int act) {
  const int lane = threadIdx.x;
  const int hf   = lane >> 4;
  const int lr   = lane & 15;
  const int row  = blockIdx.x * 16 + lr;
  const int rowc = (row < P) ? row : (P - 1);          // clamp, mask at store
  const int n0   = blockIdx.y * 32;                    // first col tile
  const int n1o  = n0 + 16;                            // second col tile
  const int n1   = (n1o + 16 <= Npad) ? n1o : (Npad - 16);  // clamp loads

  const half_t* Xrow = X + (size_t)rowc * Kpad + 16 * hf;
  const int Kt = Kpad >> 5;

  v8f acc0 = {}, acc1 = {};
  auto step = [&](int t) {
    const half_t* xp = Xrow + (size_t)t * 32;
    v8h alo = *(const v8h*)xp;
    v8h ahi = *(const v8h*)(xp + 8);
    v16h a = __builtin_shufflevector(alo, ahi, 0,1,2,3,4,5,6,7,8,9,10,11,12,13,14,15);

    const half_t* wp0 = Wp + ((size_t)t * Npad + n0 + lr) * 32 + 16 * hf;
    v8h b0l = *(const v8h*)wp0;
    v8h b0h = *(const v8h*)(wp0 + 8);
    v16h b0 = __builtin_shufflevector(b0l, b0h, 0,1,2,3,4,5,6,7,8,9,10,11,12,13,14,15);
    acc0 = __builtin_amdgcn_wmma_f32_16x16x32_f16(false, a, false, b0,
                                                  (short)0, acc0, false, false);

    const half_t* wp1 = Wp + ((size_t)t * Npad + n1 + lr) * 32 + 16 * hf;
    v8h b1l = *(const v8h*)wp1;
    v8h b1h = *(const v8h*)(wp1 + 8);
    v16h b1 = __builtin_shufflevector(b1l, b1h, 0,1,2,3,4,5,6,7,8,9,10,11,12,13,14,15);
    acc1 = __builtin_amdgcn_wmma_f32_16x16x32_f16(false, a, false, b1,
                                                  (short)0, acc1, false, false);
  };

  int t = 0;
  for (; t + 2 <= Kt; t += 2) {
    if (t + 2 < Kt) __builtin_prefetch(Xrow + (size_t)(t + 2) * 32, 0, 1);
    step(t); step(t + 1);
  }
  if (t < Kt) step(t);

  const float bc0 = bp[n0 + lr];
  const float bc1 = bp[n1 + lr];
  const bool  do1 = (n1o < Npad);
#pragma unroll
  for (int r = 0; r < 8; ++r) {
    const int m = blockIdx.x * 16 + r + 8 * hf;
    float v0 = acc0[r] + bc0;
    float v1 = acc1[r] + bc1;
    if (act == ACT_RELU)       { v0 = fmaxf(v0, 0.0f); v1 = fmaxf(v1, 0.0f); }
    else if (act == ACT_LEAKY) { v0 = (v0 > 0.0f) ? v0 : 0.01f * v0;
                                 v1 = (v1 > 0.0f) ? v1 : 0.01f * v1; }
    if (m < P) {
      if (Yf) {
        if (n0 + lr < Cout) Yf[(size_t)m * Cout + n0 + lr] = v0;
        if (do1 && n1 + lr < Cout) Yf[(size_t)m * Cout + n1 + lr] = v1;
      } else {
        Yh[(size_t)m * Npad + n0 + lr] = (half_t)v0;
        if (do1) Yh[(size_t)m * Npad + n1 + lr] = (half_t)v1;
      }
    }
  }
}

// Pack W[Cin,Cout] (f32) -> Wp[Ktiles][Npad][32] (f16, B-fragment order),
// bias -> bp[Npad] (f32), zero padding everywhere.
__global__ void pack_w_kernel(const float* __restrict__ W, const float* __restrict__ b,
                              half_t* __restrict__ Wp, float* __restrict__ bp,
                              int Cin, int Cout, int Npad, int total) {
  const int id = blockIdx.x * blockDim.x + threadIdx.x;
  if (id >= total) return;
  const int t   = id / (Npad * 32);
  const int rem = id % (Npad * 32);
  const int n   = rem / 32;
  const int kk  = rem % 32;
  const int k   = t * 32 + kk;
  Wp[id] = (k < Cin && n < Cout) ? (half_t)W[(size_t)k * Cout + n] : (half_t)0.0f;
  if (id < Npad) bp[id] = (id < Cout) ? b[id] : 0.0f;
}

__global__ void zfill_h_kernel(half_t* __restrict__ p, int total) {
  const int id = blockIdx.x * blockDim.x + threadIdx.x;
  if (id < total) p[id] = (half_t)0.0f;
}

// ---------------------------------------------------------------------------
// Farthest point sampling: one block per batch, sequential like lax.scan.
// ---------------------------------------------------------------------------
__global__ void fps_kernel(const float* __restrict__ pos, int Npts, int M,
                           int* __restrict__ idx_out, float* __restrict__ pos_out) {
  const int b = blockIdx.x;
  const float* p = pos + (size_t)b * Npts * 3;
  int*   idx = idx_out + (size_t)b * M;
  float* po  = pos_out + (size_t)b * M * 3;

  __shared__ float dmin[FPS_MAXN];
  __shared__ float rv[BDIM];
  __shared__ int   ri[BDIM];
  __shared__ float last[3];

  const int t = threadIdx.x;
  for (int j = t; j < Npts; j += blockDim.x) dmin[j] = 1e10f;
  if (t == 0) {
    idx[0] = 0;
    last[0] = p[0]; last[1] = p[1]; last[2] = p[2];
    po[0] = p[0]; po[1] = p[1]; po[2] = p[2];
  }
  __syncthreads();

  for (int i = 1; i < M; ++i) {
    const float lx = last[0], ly = last[1], lz = last[2];
    float bv = -1.0f; int bi = 0;
    for (int j = t; j < Npts; j += blockDim.x) {
      const float dx = p[3 * j] - lx, dy = p[3 * j + 1] - ly, dz = p[3 * j + 2] - lz;
      const float dm = fminf(dmin[j], dx * dx + dy * dy + dz * dz);
      dmin[j] = dm;
      if (dm > bv) { bv = dm; bi = j; }
    }
    rv[t] = bv; ri[t] = bi;
    __syncthreads();
    for (int s = blockDim.x / 2; s > 0; s >>= 1) {
      if (t < s && rv[t + s] > rv[t]) { rv[t] = rv[t + s]; ri[t] = ri[t + s]; }
      __syncthreads();
    }
    if (t == 0) {
      const int sel = ri[0];
      idx[i] = sel;
      last[0] = p[3 * sel]; last[1] = p[3 * sel + 1]; last[2] = p[3 * sel + 2];
      po[3 * i] = last[0]; po[3 * i + 1] = last[1]; po[3 * i + 2] = last[2];
    }
    __syncthreads();
  }
}

// ---------------------------------------------------------------------------
// Radius grouping: one block per center; 64 successive block-argmins over LDS
// distances == top_k(-d2, 64); valid = (d2 <= r2).
// ---------------------------------------------------------------------------
__global__ void sa_group_kernel(const float* __restrict__ src_pos,
                                const float* __restrict__ ctr_pos,
                                int Ns, int M, float r2,
                                int* __restrict__ nbr, int* __restrict__ valid) {
  const int g = blockIdx.x;           // b*M + m
  const int b = g / M;
  const float* sp = src_pos + (size_t)b * Ns * 3;
  const float cx = ctr_pos[(size_t)g * 3 + 0];
  const float cy = ctr_pos[(size_t)g * 3 + 1];
  const float cz = ctr_pos[(size_t)g * 3 + 2];

  __shared__ float d2[FPS_MAXN];
  __shared__ float rv[BDIM];
  __shared__ int   ri[BDIM];
  const int t = threadIdx.x;

  for (int j = t; j < Ns; j += blockDim.x) {
    const float dx = sp[3 * j] - cx, dy = sp[3 * j + 1] - cy, dz = sp[3 * j + 2] - cz;
    d2[j] = dx * dx + dy * dy + dz * dz;
  }
  __syncthreads();

  for (int k = 0; k < KNBR; ++k) {
    float bv = 1e30f; int bi = 0;
    for (int j = t; j < Ns; j += blockDim.x)
      if (d2[j] < bv) { bv = d2[j]; bi = j; }
    rv[t] = bv; ri[t] = bi;
    __syncthreads();
    for (int s = blockDim.x / 2; s > 0; s >>= 1) {
      if (t < s && rv[t + s] < rv[t]) { rv[t] = rv[t + s]; ri[t] = ri[t + s]; }
      __syncthreads();
    }
    if (t == 0) {
      const int sel = ri[0];
      nbr[(size_t)g * KNBR + k]   = sel;
      valid[(size_t)g * KNBR + k] = (rv[0] <= r2) ? 1 : 0;
      d2[sel] = 1e30f;
    }
    __syncthreads();
  }
}

// Relative-position columns (3 halfs) of a grouped feature matrix.
__global__ void rel_feat_kernel(const float* __restrict__ src_pos,
                                const float* __restrict__ ctr_pos,
                                const int* __restrict__ nbr,
                                int M, int Ns, half_t* __restrict__ F,
                                int Spad, int col0, int total) {
  const int i = blockIdx.x * blockDim.x + threadIdx.x;
  if (i >= total) return;
  const int g = i >> 6;
  const int b = g / M;
  const int n = nbr[i];
  const float* sp = src_pos + ((size_t)b * Ns + n) * 3;
  const float* cp = ctr_pos + (size_t)g * 3;
  half_t* f = F + (size_t)i * Spad + col0;
  f[0] = (half_t)(sp[0] - cp[0]);
  f[1] = (half_t)(sp[1] - cp[1]);
  f[2] = (half_t)(sp[2] - cp[2]);
}

// Gathered skip-feature columns of a grouped feature matrix (f16 -> f16).
__global__ void gather_cols_kernel(const half_t* __restrict__ xsrc,
                                   const int* __restrict__ nbr,
                                   int M, int Ns, half_t* __restrict__ F,
                                   int Spad, int C, int col0, int total) {
  const int id = blockIdx.x * blockDim.x + threadIdx.x;
  if (id >= total) return;
  const int i = id / C, c = id % C;
  const int g = i >> 6;
  const int b = g / M;
  const int n = nbr[i];
  F[(size_t)i * Spad + col0 + c] = xsrc[((size_t)b * Ns + n) * C + c];
}

// Masked max over the K=64 neighbor axis (f16 in/out, f32 compare).
__global__ void masked_max_kernel(const half_t* __restrict__ H,
                                  const int* __restrict__ valid,
                                  half_t* __restrict__ out, int C, int total) {
  const int id = blockIdx.x * blockDim.x + threadIdx.x;
  if (id >= total) return;
  const int g = id / C, c = id % C;
  const half_t* hp = H + ((size_t)g * KNBR) * C + c;
  const int* vp = valid + (size_t)g * KNBR;
  float m = -1e10f;
  for (int k = 0; k < KNBR; ++k)
    if (vp[k]) m = fmaxf(m, (float)hp[(size_t)k * C]);
  out[id] = (half_t)m;
}

// Copy src[R,Cs] (f16) into columns [col0,col0+Cs) of F[R,Spad].
__global__ void copy_cols_kernel(const half_t* __restrict__ src, half_t* __restrict__ F,
                                 int Cs, int Spad, int col0, int total) {
  const int id = blockIdx.x * blockDim.x + threadIdx.x;
  if (id >= total) return;
  const int r = id / Cs, c = id % Cs;
  F[(size_t)r * Spad + col0 + c] = src[id];
}

// f32 positions -> f16 columns of a feature matrix (row stride Spad).
__global__ void poscols_kernel(const float* __restrict__ src, half_t* __restrict__ F,
                               int Spad, int col0, int total) {
  const int id = blockIdx.x * blockDim.x + threadIdx.x;
  if (id >= total) return;
  const int r = id / 3, c = id % 3;
  F[(size_t)r * Spad + col0 + c] = (half_t)src[id];
}

// Broadcast src[B,Cs] (f16) across rows_per_b rows into columns of F[R,Spad].
__global__ void bcast_rows_kernel(const half_t* __restrict__ src, half_t* __restrict__ F,
                                  int rows_per_b, int Cs, int Spad, int col0, int total) {
  const int id = blockIdx.x * blockDim.x + threadIdx.x;
  if (id >= total) return;
  const int r = id / Cs, c = id % Cs;
  const int b = r / rows_per_b;
  F[(size_t)r * Spad + col0 + c] = src[(size_t)b * Cs + c];
}

// Per-batch max pool over M2 rows of H[B*M2,1024] (f16), concat f32 latents.
__global__ void pool_concat_kernel(const half_t* __restrict__ H,
                                   const float* __restrict__ latents,
                                   half_t* __restrict__ xcat, int M2, int total) {
  const int id = blockIdx.x * blockDim.x + threadIdx.x;
  if (id >= total) return;
  const int b = id / 1152, c = id % 1152;
  if (c < 1024) {
    float m = -1e30f;
    for (int j = 0; j < M2; ++j)
      m = fmaxf(m, (float)H[((size_t)b * M2 + j) * 1024 + c]);
    xcat[id] = (half_t)m;
  } else {
    xcat[id] = (half_t)latents[(size_t)b * 128 + (c - 1024)];
  }
}

// kNN (k=3) indices + normalized 1/d2 weights (PyG knn_interpolate).
__global__ void knn3_kernel(const float* __restrict__ src_pos,
                            const float* __restrict__ dst_pos,
                            int Ns, int Nd, int total,
                            int* __restrict__ knn_idx, float* __restrict__ knn_w) {
  const int i = blockIdx.x * blockDim.x + threadIdx.x;
  if (i >= total) return;
  const int b = i / Nd;
  const float* sp = src_pos + (size_t)b * Ns * 3;
  const float px = dst_pos[(size_t)i * 3 + 0];
  const float py = dst_pos[(size_t)i * 3 + 1];
  const float pz = dst_pos[(size_t)i * 3 + 2];
  float d0 = 1e30f, d1 = 1e30f, d2v = 1e30f;
  int i0 = 0, i1 = 0, i2 = 0;
  for (int j = 0; j < Ns; ++j) {
    const float dx = sp[3 * j] - px, dy = sp[3 * j + 1] - py, dz = sp[3 * j + 2] - pz;
    const float d = dx * dx + dy * dy + dz * dz;
    if (d < d0)      { d2v = d1; i2 = i1; d1 = d0; i1 = i0; d0 = d; i0 = j; }
    else if (d < d1) { d2v = d1; i2 = i1; d1 = d; i1 = j; }
    else if (d < d2v){ d2v = d; i2 = j; }
  }
  float w0 = 1.0f / fmaxf(d0, 1e-16f);
  float w1 = 1.0f / fmaxf(d1, 1e-16f);
  float w2 = 1.0f / fmaxf(d2v, 1e-16f);
  const float s = w0 + w1 + w2;
  knn_idx[3 * i + 0] = b * Ns + i0;
  knn_idx[3 * i + 1] = b * Ns + i1;
  knn_idx[3 * i + 2] = b * Ns + i2;
  knn_w[3 * i + 0] = w0 / s;
  knn_w[3 * i + 1] = w1 / s;
  knn_w[3 * i + 2] = w2 / s;
}

// Weighted gather-sum of 3 source rows (f16) into columns of F (f16).
__global__ void interp_kernel(const half_t* __restrict__ xsrc,
                              const int* __restrict__ knn_idx,
                              const float* __restrict__ knn_w,
                              half_t* __restrict__ F, int C, int Spad, int col0, int total) {
  const int id = blockIdx.x * blockDim.x + threadIdx.x;
  if (id >= total) return;
  const int r = id / C, c = id % C;
  const int*   ip = knn_idx + 3 * r;
  const float* wp = knn_w   + 3 * r;
  const float v = wp[0] * (float)xsrc[(size_t)ip[0] * C + c] +
                  wp[1] * (float)xsrc[(size_t)ip[1] * C + c] +
                  wp[2] * (float)xsrc[(size_t)ip[2] * C + c];
  F[(size_t)r * Spad + col0 + c] = (half_t)v;
}

// ---------------------------------------------------------------------------
// Host orchestration
// ---------------------------------------------------------------------------
static const int kB = 8, kN = 2048, kM1 = 410, kM2 = 103;
static const float kR1 = 0.2f, kR2 = 0.4f;

// Layer dims in params insertion order: sa1, sa2, gsa, fp3, fp2, fp1, lin1-3.
static const int kLCin[19]  = {3, 64, 64,   131, 128, 128,  259, 256, 512,
                               1408, 256,   384, 256,       128, 256, 128,
                               128, 128, 128};
static const int kLCout[19] = {64, 64, 128, 128, 128, 256,  256, 512, 1024,
                               256, 256,    256, 128,       256, 128, 128,
                               128, 128, 3};

extern "C" void kernel_launch(void* const* d_in, const int* in_sizes, int n_in,
                              void* d_out, int out_size, void* d_ws, size_t ws_size,
                              hipStream_t stream) {
  (void)in_sizes; (void)n_in; (void)out_size; (void)ws_size;
  const float* pos     = (const float*)d_in[0];   // [B,N,3]
  const float* latents = (const float*)d_in[1];   // [B,128]
  const float* Wt[19]; const float* Bs[19];
  {
    int p = 2;
    for (int i = 0; i < 19; ++i) { Wt[i] = (const float*)d_in[p++]; Bs[i] = (const float*)d_in[p++]; }
  }

  int Kp[19], Np[19];
  for (int i = 0; i < 19; ++i) {
    Kp[i] = (kLCin[i] + 31) / 32 * 32;
    Np[i] = (kLCout[i] + 15) / 16 * 16;
  }

  // Workspace carve-up, 64B-aligned slices (~115 MB total).
  char* w0 = (char*)d_ws;
  size_t off = 0;
  auto carve = [&](size_t bytes) {
    void* q = w0 + off;
    off = (off + bytes + 63) & ~(size_t)63;
    return q;
  };
  const int P1 = kB * kM1 * KNBR;   // 209920
  const int P2 = kB * kM2 * KNBR;   // 52736
  const int G1 = kB * kM1, G2 = kB * kM2;
  const int D  = kB * kN;

  int*    idx1 = (int*)   carve((size_t)G1 * 4);
  float*  pos1 = (float*) carve((size_t)G1 * 3 * 4);
  int*    idx2 = (int*)   carve((size_t)G2 * 4);
  float*  pos2 = (float*) carve((size_t)G2 * 3 * 4);
  int*    nbr1 = (int*)   carve((size_t)P1 * 4);
  int*    val1 = (int*)   carve((size_t)P1 * 4);
  int*    nbr2 = (int*)   carve((size_t)P2 * 4);
  int*    val2 = (int*)   carve((size_t)P2 * 4);
  half_t* x1   = (half_t*)carve((size_t)G1 * 128 * 2);
  half_t* x2   = (half_t*)carve((size_t)G2 * 256 * 2);
  half_t* x3c  = (half_t*)carve((size_t)kB * 1152 * 2);
  half_t* xfp3 = (half_t*)carve((size_t)G2 * 256 * 2);
  half_t* xfp2 = (half_t*)carve((size_t)G1 * 128 * 2);
  int*    kidx = (int*)   carve((size_t)D * 3 * 4);
  float*  kw   = (float*) carve((size_t)D * 3 * 4);
  half_t* Wp[19]; float* Bp[19];
  for (int i = 0; i < 19; ++i) {
    Wp[i] = (half_t*)carve((size_t)Kp[i] * Np[i] * 2);
    Bp[i] = (float*) carve((size_t)Np[i] * 4);
  }
  half_t* bufA = (half_t*)carve((size_t)P1 * 128 * 2);
  half_t* bufB = (half_t*)carve((size_t)P1 * 128 * 2);
  (void)idx1; (void)idx2;

  auto blk = [](long total) { return dim3((unsigned)((total + BDIM - 1) / BDIM)); };
  auto gemm = [&](const half_t* X, int li, half_t* Yh, float* Yf, int P, int act) {
    dim3 grid((P + 15) / 16, (Np[li] + 31) / 32);
    wmma_gemm_kernel<<<grid, 32, 0, stream>>>(X, Wp[li], Bp[li], Yh, Yf,
                                              P, Kp[li], Np[li], kLCout[li], act);
  };

  // ---- Pack all weights to f16 fragment order --------------------------
  for (int i = 0; i < 19; ++i) {
    const int total = Kp[i] * Np[i];
    pack_w_kernel<<<blk(total), BDIM, 0, stream>>>(Wt[i], Bs[i], Wp[i], Bp[i],
                                                   kLCin[i], kLCout[i], Np[i], total);
  }

  // ---- SA1 -------------------------------------------------------------
  fps_kernel<<<kB, BDIM, 0, stream>>>(pos, kN, kM1, idx1, pos1);
  sa_group_kernel<<<G1, BDIM, 0, stream>>>(pos, pos1, kN, kM1, kR1 * kR1, nbr1, val1);
  zfill_h_kernel<<<blk((long)P1 * 32), BDIM, 0, stream>>>(bufA, P1 * 32);
  rel_feat_kernel<<<blk(P1), BDIM, 0, stream>>>(pos, pos1, nbr1, kM1, kN, bufA, 32, 0, P1);
  gemm(bufA, 0, bufB, nullptr, P1, ACT_RELU);              // 3 -> 64
  gemm(bufB, 1, bufA, nullptr, P1, ACT_RELU);              // 64 -> 64
  gemm(bufA, 2, bufB, nullptr, P1, ACT_RELU);              // 64 -> 128
  masked_max_kernel<<<blk((long)G1 * 128), BDIM, 0, stream>>>(bufB, val1, x1, 128, G1 * 128);

  // ---- SA2 -------------------------------------------------------------
  fps_kernel<<<kB, BDIM, 0, stream>>>(pos1, kM1, kM2, idx2, pos2);
  sa_group_kernel<<<G2, BDIM, 0, stream>>>(pos1, pos2, kM1, kM2, kR2 * kR2, nbr2, val2);
  zfill_h_kernel<<<blk((long)P2 * 160), BDIM, 0, stream>>>(bufA, P2 * 160);
  gather_cols_kernel<<<blk((long)P2 * 128), BDIM, 0, stream>>>(x1, nbr2, kM2, kM1, bufA, 160, 128, 0, P2 * 128);
  rel_feat_kernel<<<blk(P2), BDIM, 0, stream>>>(pos1, pos2, nbr2, kM2, kM1, bufA, 160, 128, P2);
  gemm(bufA, 3, bufB, nullptr, P2, ACT_RELU);              // 131 -> 128
  gemm(bufB, 4, bufA, nullptr, P2, ACT_RELU);              // 128 -> 128
  gemm(bufA, 5, bufB, nullptr, P2, ACT_RELU);              // 128 -> 256
  masked_max_kernel<<<blk((long)G2 * 256), BDIM, 0, stream>>>(bufB, val2, x2, 256, G2 * 256);

  // ---- Global SA + latent concat --------------------------------------
  zfill_h_kernel<<<blk((long)G2 * 288), BDIM, 0, stream>>>(bufA, G2 * 288);
  copy_cols_kernel<<<blk((long)G2 * 256), BDIM, 0, stream>>>(x2, bufA, 256, 288, 0, G2 * 256);
  poscols_kernel<<<blk((long)G2 * 3), BDIM, 0, stream>>>(pos2, bufA, 288, 256, G2 * 3);
  gemm(bufA, 6, bufB, nullptr, G2, ACT_RELU);              // 259 -> 256
  gemm(bufB, 7, bufA, nullptr, G2, ACT_RELU);              // 256 -> 512
  gemm(bufA, 8, bufB, nullptr, G2, ACT_RELU);              // 512 -> 1024
  pool_concat_kernel<<<blk((long)kB * 1152), BDIM, 0, stream>>>(bufB, latents, x3c, kM2, kB * 1152);

  // ---- FP3 (k=1: broadcast global feature) -----------------------------
  bcast_rows_kernel<<<blk((long)G2 * 1152), BDIM, 0, stream>>>(x3c, bufA, kM2, 1152, 1408, 0, G2 * 1152);
  copy_cols_kernel<<<blk((long)G2 * 256), BDIM, 0, stream>>>(x2, bufA, 256, 1408, 1152, G2 * 256);
  gemm(bufA, 9, bufB, nullptr, G2, ACT_RELU);              // 1408 -> 256
  gemm(bufB, 10, xfp3, nullptr, G2, ACT_RELU);             // 256 -> 256

  // ---- FP2 (k=3 pos2 -> pos1, skip x1) ---------------------------------
  knn3_kernel<<<blk(G1), BDIM, 0, stream>>>(pos2, pos1, kM2, kM1, G1, kidx, kw);
  interp_kernel<<<blk((long)G1 * 256), BDIM, 0, stream>>>(xfp3, kidx, kw, bufA, 256, 384, 0, G1 * 256);
  copy_cols_kernel<<<blk((long)G1 * 128), BDIM, 0, stream>>>(x1, bufA, 128, 384, 256, G1 * 128);
  gemm(bufA, 11, bufB, nullptr, G1, ACT_RELU);             // 384 -> 256
  gemm(bufB, 12, xfp2, nullptr, G1, ACT_RELU);             // 256 -> 128

  // ---- FP1 (k=3 pos1 -> pos, no skip) ----------------------------------
  knn3_kernel<<<blk(D), BDIM, 0, stream>>>(pos1, pos, kM1, kN, D, kidx, kw);
  interp_kernel<<<blk((long)D * 128), BDIM, 0, stream>>>(xfp2, kidx, kw, bufA, 128, 128, 0, D * 128);
  gemm(bufA, 13, bufB, nullptr, D, ACT_RELU);              // 128 -> 256
  gemm(bufB, 14, bufA, nullptr, D, ACT_RELU);              // 256 -> 128
  gemm(bufA, 15, bufB, nullptr, D, ACT_RELU);              // 128 -> 128

  // ---- Head ------------------------------------------------------------
  gemm(bufB, 16, bufA, nullptr, D, ACT_LEAKY);             // 128 -> 128
  gemm(bufA, 17, bufB, nullptr, D, ACT_LEAKY);             // 128 -> 128
  gemm(bufB, 18, nullptr, (float*)d_out, D, ACT_NONE);     // 128 -> 3
}
